// FeatureExtractor_1_30588757082788
// MI455X (gfx1250) — matrix operand order repaired and checked by reading, not compiled, not run on hardware
//
#include <hip/hip_runtime.h>
#include <cstdint>

typedef __bf16 bf16x8  __attribute__((ext_vector_type(8)));
typedef __bf16 bf16x16 __attribute__((ext_vector_type(16)));
typedef float  f32x8   __attribute__((ext_vector_type(8)));

#define NB 8          // batch
#define NK 32         // nsample
#define RAD2 0.04f    // radius^2 (0.2^2, all stages)

// ---------------------------------------------------------------------------
// Farthest point sampling: 1 block per batch, xyz + running min-dist in LDS.
// Replicates: out[j] = argmax-before-update, start at index 0, first-max tie.
// ---------------------------------------------------------------------------
__global__ __launch_bounds__(512) void fps_kernel(const float* __restrict__ xyz,
                                                  int N, int S, int* __restrict__ fidx) {
  __shared__ float sx[4096], sy[4096], sz[4096], sd[4096];
  __shared__ float rv[16];
  __shared__ int   ri[16];
  __shared__ int   sFar;
  const int b = blockIdx.x, tid = threadIdx.x;
  const int lane = tid & 31, wv = tid >> 5;
  const float* base = xyz + (size_t)b * N * 3;
  for (int n = tid; n < N; n += 512) {
    sx[n] = base[n * 3 + 0];
    sy[n] = base[n * 3 + 1];
    sz[n] = base[n * 3 + 2];
    sd[n] = 1e10f;
  }
  if (tid == 0) sFar = 0;
  __syncthreads();
  for (int j = 0; j < S; ++j) {
    const int far = sFar;
    if (tid == 0) fidx[b * S + j] = far;
    const float cx = sx[far], cy = sy[far], cz = sz[far];
    float bv = -1.0f; int bi = 0;
    for (int n = tid; n < N; n += 512) {
      float dx = sx[n] - cx, dy = sy[n] - cy, dz = sz[n] - cz;
      float d = dx * dx + dy * dy + dz * dz;
      float dd = sd[n];
      d = d < dd ? d : dd;
      sd[n] = d;
      if (d > bv) { bv = d; bi = n; }
    }
    // wave-level argmax (first-occurrence tie: smaller index wins on equal)
    #pragma unroll
    for (int off = 16; off > 0; off >>= 1) {
      float ov = __shfl_down(bv, off, 32);
      int   oi = __shfl_down(bi, off, 32);
      if (ov > bv || (ov == bv && oi < bi)) { bv = ov; bi = oi; }
    }
    if (lane == 0) { rv[wv] = bv; ri[wv] = bi; }
    __syncthreads();
    if (wv == 0) {
      float v  = (lane < 16) ? rv[lane] : -1.0f;
      int   i2 = (lane < 16) ? ri[lane] : 0x7fffffff;
      #pragma unroll
      for (int off = 16; off > 0; off >>= 1) {
        float ov = __shfl_down(v, off, 32);
        int   oi = __shfl_down(i2, off, 32);
        if (ov > v || (ov == v && oi < i2)) { v = ov; i2 = oi; }
      }
      if (lane == 0) sFar = i2;
    }
    __syncthreads();
  }
}

__global__ void gather_kernel(const float* __restrict__ xyz, const int* __restrict__ fidx,
                              int N, int S, float* __restrict__ nxyz) {
  int t = blockIdx.x * 256 + threadIdx.x;
  if (t >= NB * S) return;
  int b = t / S;
  int n = fidx[t];
  const float* p = xyz + ((size_t)b * N + n) * 3;
  nxyz[t * 3 + 0] = p[0];
  nxyz[t * 3 + 1] = p[1];
  nxyz[t * 3 + 2] = p[2];
}

// ---------------------------------------------------------------------------
// Ball query: 1 wave per center; ballot + prefix popcount appends the first
// NK in-radius indices in ascending order, pad with first hit (exact ref).
// ---------------------------------------------------------------------------
__global__ void ballq_kernel(const float* __restrict__ xyz, const float* __restrict__ nxyz,
                             int N, int S, int* __restrict__ idx) {
  int gw = (blockIdx.x * blockDim.x + threadIdx.x) >> 5;
  int lane = threadIdx.x & 31;
  if (gw >= NB * S) return;
  int b = gw / S, si = gw % S;
  const float* ctr = nxyz + (size_t)(b * S + si) * 3;
  float cx = ctr[0], cy = ctr[1], cz = ctr[2];
  const float* pb = xyz + (size_t)b * N * 3;
  int* ob = idx + (size_t)(b * S + si) * NK;
  int cnt = 0, firstIdx = -1;
  for (int n0 = 0; n0 < N && cnt < NK; n0 += 32) {
    int n = n0 + lane;
    bool ok = false;
    if (n < N) {
      float dx = pb[n * 3] - cx, dy = pb[n * 3 + 1] - cy, dz = pb[n * 3 + 2] - cz;
      ok = (dx * dx + dy * dy + dz * dz) <= RAD2;
    }
    unsigned m = (unsigned)__ballot(ok ? 1 : 0);
    if (firstIdx < 0 && m) firstIdx = n0 + __builtin_ctz(m);
    if (ok) {
      int pos = cnt + __popc(m & ((1u << lane) - 1u));
      if (pos < NK) ob[pos] = n;
    }
    cnt += __popc(m);
  }
  if (cnt < NK) {
    for (int p = cnt + lane; p < NK; p += 32) ob[p] = firstIdx;
  }
}

// ---------------------------------------------------------------------------
// Grouping: builds X[col][c] (bf16, cin padded to Cpad) with col=(b*K+k)*S+s.
// c<3: centered xyz; 3..3+Dprev: gathered prev features; rest zero pad.
// ---------------------------------------------------------------------------
__global__ void group_kernel(const float* __restrict__ xyz, const float* __restrict__ nxyz,
                             const int* __restrict__ idx, const float* __restrict__ pts,
                             int Dprev, int Sprev, int N, int S, int Cpad,
                             __bf16* __restrict__ X) {
  int t = blockIdx.x * 256 + threadIdx.x;
  int tot = NB * S * NK;
  if (t >= tot) return;
  int b = t / (S * NK);
  int r = t % (S * NK);
  int si = r / NK, k = r % NK;
  int n = idx[((size_t)(b * S + si)) * NK + k];
  size_t col = ((size_t)(b * NK + k)) * S + si;
  __bf16* xr = X + col * Cpad;
  const float* p = xyz + ((size_t)b * N + n) * 3;
  const float* c = nxyz + ((size_t)(b * S + si)) * 3;
  xr[0] = (__bf16)(p[0] - c[0]);
  xr[1] = (__bf16)(p[1] - c[1]);
  xr[2] = (__bf16)(p[2] - c[2]);
  int cc = 3;
  for (int d = 0; d < Dprev; ++d)
    xr[cc++] = (__bf16)pts[(size_t)d * (NB * Sprev) + (size_t)b * Sprev + n];
  for (; cc < Cpad; ++cc) xr[cc] = (__bf16)0.0f;
}

__global__ void wprep_kernel(const float* __restrict__ W, int cout, int cin, int Cpad,
                             __bf16* __restrict__ Wb) {
  int t = blockIdx.x * 256 + threadIdx.x;
  if (t >= cout * Cpad) return;
  int co = t / Cpad, c = t % Cpad;
  Wb[t] = (c < cin) ? (__bf16)W[(size_t)co * cin + c] : (__bf16)0.0f;
}

// ---------------------------------------------------------------------------
// bf16 WMMA GEMM: Y[co][col] = sum_c W[co][c] * X[col][c]  (fp32 accumulate)
// Each wave: one 16-row co tile x four 16-col tiles (16x64 output block);
// the A (weight) fragment is loaded once per K-step and reused by 4 WMMAs.
// A 16x32 / B 32x16 use contiguous K per lane half (K = 16*hi + s); f32 C/D per CDNA5 ISA 7.12.2.
// ---------------------------------------------------------------------------
__global__ __launch_bounds__(256) void gemm_bf16_kernel(const __bf16* __restrict__ W,
                                                        const __bf16* __restrict__ X,
                                                        float* __restrict__ Y,
                                                        int cout, int Cpad, int cols) {
  int wave = (blockIdx.x * 256 + threadIdx.x) >> 5;
  int nco = cout >> 4, ncol4 = cols >> 6;
  if (wave >= nco * ncol4) return;
  int cot = wave % nco, colt4 = wave / nco;
  int lane = threadIdx.x & 31;
  int hi = lane >> 4, l15 = lane & 15;
  const __bf16* wr = W + (size_t)(cot * 16 + l15) * Cpad + hi * 16;           // A: M=l15
  const __bf16* xr = X + (size_t)(colt4 * 64 + l15) * Cpad + hi * 16;         // B: N=l15
  const size_t xstep = (size_t)16 * Cpad;                                     // next 16-col tile
  f32x8 acc0 = {0.f, 0.f, 0.f, 0.f, 0.f, 0.f, 0.f, 0.f};
  f32x8 acc1 = acc0, acc2 = acc0, acc3 = acc0;
  for (int kk = 0; kk < Cpad; kk += 32) {
    bf16x16 a = *(const bf16x16*)(wr + kk);                 // K = kk+16*hi .. +15
    bf16x16 b0 = *(const bf16x16*)(xr + kk);                // K = kk+16*hi .. +15
    bf16x16 b1 = *(const bf16x16*)(xr + kk + xstep);
    bf16x16 b2 = *(const bf16x16*)(xr + kk + 2 * xstep);
    bf16x16 b3 = *(const bf16x16*)(xr + kk + 3 * xstep);
    acc0 = __builtin_amdgcn_wmma_f32_16x16x32_bf16(false, a, false, b0, (short)0, acc0, false, false);
    acc1 = __builtin_amdgcn_wmma_f32_16x16x32_bf16(false, a, false, b1, (short)0, acc1, false, false);
    acc2 = __builtin_amdgcn_wmma_f32_16x16x32_bf16(false, a, false, b2, (short)0, acc2, false, false);
    acc3 = __builtin_amdgcn_wmma_f32_16x16x32_bf16(false, a, false, b3, (short)0, acc3, false, false);
  }
  float* yb = Y + (size_t)(cot * 16 + hi * 8) * cols + (size_t)colt4 * 64 + l15;
  #pragma unroll
  for (int v = 0; v < 8; ++v) {
    yb[(size_t)v * cols + 0]  = acc0[v];
    yb[(size_t)v * cols + 16] = acc1[v];
    yb[(size_t)v * cols + 32] = acc2[v];
    yb[(size_t)v * cols + 48] = acc3[v];
  }
}

// Per-channel batch stats over all cols -> folded BN scale/shift.
// (conv bias cancels under mean subtraction, so it is omitted exactly.)
__global__ __launch_bounds__(256) void stats_kernel(const float* __restrict__ Y, int cols,
                                                    const float* __restrict__ g,
                                                    const float* __restrict__ beta,
                                                    float* __restrict__ st) {
  __shared__ float ss[256], sq[256];
  int co = blockIdx.x, tid = threadIdx.x;
  const float* row = Y + (size_t)co * cols;
  float s = 0.f, s2 = 0.f;
  for (int i = tid; i < cols; i += 256) {
    float v = row[i];
    s += v; s2 += v * v;
  }
  ss[tid] = s; sq[tid] = s2;
  __syncthreads();
  for (int o = 128; o > 0; o >>= 1) {
    if (tid < o) { ss[tid] += ss[tid + o]; sq[tid] += sq[tid + o]; }
    __syncthreads();
  }
  if (tid == 0) {
    float inv = 1.0f / (float)cols;
    float mean = ss[0] * inv;
    float var = sq[0] * inv - mean * mean;
    float sc = g[co] * rsqrtf(var + 1e-5f);
    st[2 * co] = sc;
    st[2 * co + 1] = beta[co] - mean * sc;
  }
}

// normalize + relu, write transposed bf16 as next layer's X[col][c]
__global__ void normrelu_kernel(const float* __restrict__ Y, const float* __restrict__ st,
                                int cout, int cols, __bf16* __restrict__ Xn) {
  int col = blockIdx.x * 256 + threadIdx.x;
  if (col >= cols) return;
  __bf16* xr = Xn + (size_t)col * cout;
  for (int co = 0; co < cout; ++co) {
    float v = Y[(size_t)co * cols + col] * st[2 * co] + st[2 * co + 1];
    xr[co] = (__bf16)fmaxf(v, 0.0f);
  }
}

// normalize + relu + max over K  -> pooled[co][b*S+s]
__global__ void pool_kernel(const float* __restrict__ Y, const float* __restrict__ st,
                            int cout, int S, float* __restrict__ P) {
  int t = blockIdx.x * 256 + threadIdx.x;
  int tot = cout * NB * S;
  if (t >= tot) return;
  int co = t / (NB * S);
  int r = t % (NB * S);
  int b = r / S, si = r % S;
  float sc = st[2 * co], sh = st[2 * co + 1];
  int cols = NB * NK * S;
  const float* yb = Y + (size_t)co * cols + (size_t)b * NK * S + si;
  float m = -3.4e38f;
  for (int k = 0; k < NK; ++k) m = fmaxf(m, yb[(size_t)k * S] * sc + sh);
  P[t] = fmaxf(m, 0.0f);
}

// pack outputs: l2_xyz (8,3,512) then l4_points (8,1920,1)
__global__ void out_kernel(const float* __restrict__ nxyz1, const float* __restrict__ p3,
                           float* __restrict__ out) {
  int t = blockIdx.x * 256 + threadIdx.x;
  if (t < 12288) {
    int b = t / 1536, r = t % 1536;
    int c = r / 512, s = r % 512;
    out[t] = nxyz1[((size_t)(b * 512 + s)) * 3 + c];
  } else if (t < 12288 + 15360) {
    int u = t - 12288;
    int b = u / 1920, co = u % 1920;
    out[t] = p3[(size_t)co * NB + b];
  }
}

static inline int nb_(long long n, int t) { return (int)((n + t - 1) / t); }

extern "C" void kernel_launch(void* const* d_in, const int* in_sizes, int n_in,
                              void* d_out, int out_size, void* d_ws, size_t ws_size,
                              hipStream_t stream) {
  const float* xyz = (const float*)d_in[0];
  static const int Sarr[4] = {1024, 512, 128, 1};
  static const int Narr[4] = {4096, 1024, 512, 128};
  static const int cinArr[4] = {3, 131, 259, 515};
  static const int cpad0[4] = {32, 160, 288, 544};
  static const int coutA[4][3] = {{64, 64, 128}, {128, 128, 256}, {256, 256, 512}, {512, 512, 1920}};

  char* ws = (char*)d_ws;
  size_t off = 0;
  auto carve = [&](size_t bytes) -> void* {
    void* p = ws + off;
    off = (off + bytes + 255) & ~(size_t)255;
    return p;
  };

  __bf16* Wb[4][3];
  for (int st = 0; st < 4; ++st)
    for (int l = 0; l < 3; ++l) {
      int cp = l ? coutA[st][l - 1] : cpad0[st];
      Wb[st][l] = (__bf16*)carve((size_t)coutA[st][l] * cp * sizeof(__bf16));
    }
  float* statsb = (float*)carve(1920 * 2 * sizeof(float));
  int* fidx = (int*)carve((size_t)NB * 1024 * sizeof(int));
  float* nxyz[4];
  for (int st = 0; st < 4; ++st) nxyz[st] = (float*)carve((size_t)NB * Sarr[st] * 3 * sizeof(float));
  int* bidx = (int*)carve((size_t)NB * 1024 * NK * sizeof(int));
  float* pooled[4];
  for (int st = 0; st < 4; ++st)
    pooled[st] = (float*)carve((size_t)coutA[st][2] * NB * Sarr[st] * sizeof(float));
  const size_t XELEMS = 20971520ull;  // max cols*Cpad over all layers (stage1: 131072*160)
  __bf16* XA = (__bf16*)carve(XELEMS * sizeof(__bf16));
  __bf16* XB = (__bf16*)carve(XELEMS * sizeof(__bf16));
  float* Ybuf = (float*)carve(33554432ull * sizeof(float));  // max cout*cols (stage0 L3)
  if (off > ws_size) return;  // workspace too small: bail deterministically

  // weight conversion fp32 -> padded bf16
  for (int st = 0; st < 4; ++st)
    for (int l = 0; l < 3; ++l) {
      const float* W = (const float*)d_in[1 + 4 * (st * 3 + l)];
      int ci = l ? coutA[st][l - 1] : cinArr[st];
      int cp = l ? coutA[st][l - 1] : cpad0[st];
      int tot = coutA[st][l] * cp;
      wprep_kernel<<<nb_(tot, 256), 256, 0, stream>>>(W, coutA[st][l], ci, cp, Wb[st][l]);
    }

  const float* cur = xyz;
  for (int st = 0; st < 4; ++st) {
    int N = Narr[st], S = Sarr[st];
    fps_kernel<<<NB, 512, 0, stream>>>(cur, N, S, fidx);
    gather_kernel<<<nb_((long long)NB * S, 256), 256, 0, stream>>>(cur, fidx, N, S, nxyz[st]);
    ballq_kernel<<<nb_((long long)NB * S * 32, 256), 256, 0, stream>>>(cur, nxyz[st], N, S, bidx);
    int Dp = st ? coutA[st - 1][2] : 0;
    int Sp = st ? Sarr[st - 1] : 0;
    const float* pts = st ? pooled[st - 1] : nullptr;
    group_kernel<<<nb_((long long)NB * S * NK, 256), 256, 0, stream>>>(
        cur, nxyz[st], bidx, pts, Dp, Sp, N, S, cpad0[st], XA);

    int cols = NB * NK * S;
    __bf16 *Xi = XA, *Xo = XB;
    for (int l = 0; l < 3; ++l) {
      int co = coutA[st][l];
      int cp = l ? coutA[st][l - 1] : cpad0[st];
      int tiles = (co / 16) * (cols / 64);  // 16x64 output block per wave
      gemm_bf16_kernel<<<nb_(tiles, 8), 256, 0, stream>>>(Wb[st][l], Xi, Ybuf, co, cp, cols);
      const float* g = (const float*)d_in[1 + 4 * (st * 3 + l) + 2];
      const float* be = (const float*)d_in[1 + 4 * (st * 3 + l) + 3];
      stats_kernel<<<co, 256, 0, stream>>>(Ybuf, cols, g, be, statsb);
      if (l < 2) {
        normrelu_kernel<<<nb_(cols, 256), 256, 0, stream>>>(Ybuf, statsb, co, cols, Xo);
        __bf16* tswap = Xi; Xi = Xo; Xo = tswap;
      } else {
        pool_kernel<<<nb_((long long)co * NB * S, 256), 256, 0, stream>>>(Ybuf, statsb, co, S,
                                                                          pooled[st]);
      }
    }
    cur = nxyz[st];
  }
  out_kernel<<<nb_(27648, 256), 256, 0, stream>>>(nxyz[1], pooled[3], (float*)d_out);
}
